// SelfAttention_31198642438353
// MI455X (gfx1250) — matrix operand order repaired and checked by reading, not compiled, not run on hardware
//
#include <hip/hip_runtime.h>
#include <hip/hip_bf16.h>

typedef __attribute__((ext_vector_type(16))) _Float16 v16h;
typedef __attribute__((ext_vector_type(8)))  _Float16 v8h;
typedef __attribute__((ext_vector_type(8)))  float    v8f;
typedef __attribute__((ext_vector_type(4)))  float    v4f;

#define BB 8
#define SS 2048
#define EE 1024
#define DD 128
#define NEG_BIG (-1e30f)

// ---------------------------------------------------------------------------
// Kernel 1: QKV projection. One wave -> one 16x128 row-block of ONE output
// matrix (Q, K or V): 8 f32 accumulators, A operand loaded once per K-chunk
// and reused by 8 WMMAs (X re-read 3x total instead of 24x).
// Q: f16 row-major [B*S][128], pre-scaled by sqrt(128).
// K: f16 row-major [B*S][128].
// V: f16 TRANSPOSED [B][128][S]  (so P.V B-operand loads are contiguous).
// ---------------------------------------------------------------------------
__global__ void qkv_project_kernel(const float* __restrict__ X,
                                   const float* __restrict__ Wq,
                                   const float* __restrict__ Wk,
                                   const float* __restrict__ Wv,
                                   _Float16* __restrict__ Qh,
                                   _Float16* __restrict__ Kh,
                                   _Float16* __restrict__ Vt) {
    const int lane = threadIdx.x & 31;
    const int hi   = lane >> 4;          // half-wave select
    const int ln   = lane & 15;
    const int wid  = threadIdx.x >> 5;

    const int tile  = blockIdx.x * 8 + wid;     // 3072 wave-tiles total
    const int mtile = tile / 3;                 // 1024 row-blocks
    const int which = tile % 3;                 // 0=Q 1=K 2=V

    const float* __restrict__ W = (which == 0) ? Wq : ((which == 1) ? Wk : Wv);

    const int arow = mtile * 16 + ln;           // A row (both half-waves same M)
    const float* __restrict__ xrow = X + (size_t)arow * EE;

    v8f acc[8];
    {
        const v8f z = {};
#pragma unroll
        for (int n = 0; n < 8; ++n) acc[n] = z;
    }

    for (int kc = 0; kc < EE; kc += 32) {
        // ---- A operand: 16x32 f16. Lane half selects K sub-range:
        //  lanes 0-15 : elements 0..15 -> K=kc+0..15
        //  lanes 16-31: elements 0..15 -> K=kc+16..31
        v16h a;
        {
            const int k0 = kc + hi * 16;
            v4f f0 = *(const v4f*)(xrow + k0);
            v4f f1 = *(const v4f*)(xrow + k0 + 4);
            v4f f2 = *(const v4f*)(xrow + k0 + 8);
            v4f f3 = *(const v4f*)(xrow + k0 + 12);
#pragma unroll
            for (int e = 0; e < 4; ++e) {
                a[e]      = (_Float16)f0[e];
                a[e + 4]  = (_Float16)f1[e];
                a[e + 8]  = (_Float16)f2[e];
                a[e + 12] = (_Float16)f3[e];
            }
        }
        // ---- 8 B operands (32x16 f16 each): lane = column, lanes 0-15 hold
        //      K=kc..kc+15, lanes 16-31 K=kc+16..31 (contiguous per lane).
#pragma unroll
        for (int n = 0; n < 8; ++n) {
            const float* __restrict__ wrow = W + (size_t)(n * 16 + ln) * EE;
            v16h bmat;
            {
                const int k0 = kc + hi * 16;
                v4f g0 = *(const v4f*)(wrow + k0);
                v4f g1 = *(const v4f*)(wrow + k0 + 4);
                v4f g2 = *(const v4f*)(wrow + k0 + 8);
                v4f g3 = *(const v4f*)(wrow + k0 + 12);
#pragma unroll
                for (int e = 0; e < 4; ++e) {
                    bmat[e]      = (_Float16)g0[e];
                    bmat[e + 4]  = (_Float16)g1[e];
                    bmat[e + 8]  = (_Float16)g2[e];
                    bmat[e + 12] = (_Float16)g3[e];
                }
            }
            acc[n] = __builtin_amdgcn_wmma_f32_16x16x32_f16(
                         false, a, false, bmat, (short)0, acc[n], false, false);
        }
    }

    // ---- Store. C layout: VGPR r -> row (mtile*16 + r + hi*8), col = n*16+ln.
    if (which == 2) {
        // V transposed: Vt[(b*128 + col)*S + s]; rows r=0..7 are consecutive s.
        const int row0 = mtile * 16 + hi * 8;
        const int b    = row0 / SS;
        const int s0   = row0 % SS;
#pragma unroll
        for (int n = 0; n < 8; ++n) {
            const int bcol = n * 16 + ln;
            v8h pk;
#pragma unroll
            for (int r = 0; r < 8; ++r) pk[r] = (_Float16)acc[n][r];
            *(v8h*)(Vt + ((size_t)b * DD + bcol) * SS + s0) = pk;
        }
    } else {
        _Float16* __restrict__ Out = (which == 0) ? Qh : Kh;
        const float scale = (which == 0) ? 11.313708498984760f : 1.0f; // sqrt(128)
#pragma unroll
        for (int n = 0; n < 8; ++n) {
            const int bcol = n * 16 + ln;
#pragma unroll
            for (int r = 0; r < 8; ++r) {
                const int row = mtile * 16 + r + hi * 8;
                Out[(size_t)row * DD + bcol] = (_Float16)(acc[n][r] * scale);
            }
        }
    }
}

// ---------------------------------------------------------------------------
// Kernel 2: causal flash attention. One wave -> 16 query rows, streaming the
// key dimension in blocks of 32 with online softmax. Longest tiles first.
// ---------------------------------------------------------------------------
__global__ void flash_attn_kernel(const _Float16* __restrict__ Qh,
                                  const _Float16* __restrict__ Kh,
                                  const _Float16* __restrict__ Vt,
                                  float* __restrict__ out) {
    __shared__ __align__(16) _Float16 pbuf[8][16 * 32];   // P bounce, 1KB/wave

    const int lane = threadIdx.x & 31;
    const int hi   = lane >> 4;
    const int ln   = lane & 15;
    const int wid  = threadIdx.x >> 5;

    const int tile  = blockIdx.x * 8 + wid;   // 1024 tiles = B * (S/16)
    const int b     = tile >> 7;
    const int qt    = 127 - (tile & 127);     // descending: big tiles start first
    const int qbase = qt * 16;

    // ---- Q tile -> A-layout registers (4 chunks of K=32)
    v16h aq[4];
    {
        const _Float16* __restrict__ qrow =
            Qh + ((size_t)(b * SS + qbase + ln)) * DD;
#pragma unroll
        for (int c = 0; c < 4; ++c) {
            const int k0 = c * 32 + hi * 16;
            v8h q0 = *(const v8h*)(qrow + k0);
            v8h q1 = *(const v8h*)(qrow + k0 + 8);
#pragma unroll
            for (int e = 0; e < 8; ++e) { aq[c][e] = q0[e]; aq[c][8 + e] = q1[e]; }
        }
    }

    float m[8], l[8];
    v8f   o[8];
    {
        const v8f z = {};
#pragma unroll
        for (int r = 0; r < 8; ++r) { m[r] = NEG_BIG; l[r] = 0.0f; }
#pragma unroll
        for (int vt = 0; vt < 8; ++vt) o[vt] = z;
    }

    const _Float16* __restrict__ Kb = Kh + (size_t)b * SS * DD;
    const _Float16* __restrict__ Vb = Vt + (size_t)b * DD * SS;
    _Float16* __restrict__ pw = &pbuf[wid][0];

    for (int kb = 0; kb <= qbase + 15; kb += 32) {
        // speculative prefetch of next key block (dropped silently at the edge)
        __builtin_prefetch(Kb + (size_t)(kb + 32 + ln) * DD, 0, 0);

        // ---- scores: two 16x16 tiles (keys kb..kb+15 and kb+16..kb+31)
        v8f s[2];
#pragma unroll
        for (int t = 0; t < 2; ++t) {
            v8f sa = {};
            const _Float16* __restrict__ krow =
                Kb + (size_t)(kb + t * 16 + ln) * DD;
#pragma unroll
            for (int c = 0; c < 4; ++c) {
                // 32 contiguous bytes per lane -> direct v16h (two b128 loads)
                v16h bk = *(const v16h*)(krow + c * 32 + hi * 16);
                sa = __builtin_amdgcn_wmma_f32_16x16x32_f16(
                         false, aq[c], false, bk, (short)0, sa, false, false);
            }
            s[t] = sa;
        }

        // ---- causal mask + per-lane row max
        float mx[8];
#pragma unroll
        for (int r = 0; r < 8; ++r) {
            const int row  = qbase + r + hi * 8;
            const float v0 = ((kb + ln)      <= row) ? s[0][r] : NEG_BIG;
            const float v1 = ((kb + 16 + ln) <= row) ? s[1][r] : NEG_BIG;
            s[0][r] = v0; s[1][r] = v1;
            mx[r] = fmaxf(v0, v1);
        }
        // reduce across the 16 lanes of each half-wave (xor 1,2,4,8 stays in-group)
#pragma unroll
        for (int off = 1; off < 16; off <<= 1)
#pragma unroll
            for (int r = 0; r < 8; ++r)
                mx[r] = fmaxf(mx[r], __shfl_xor(mx[r], off, 32));

        // ---- online softmax update
        float corr[8], psum[8];
#pragma unroll
        for (int r = 0; r < 8; ++r) {
            const float mn = fmaxf(m[r], mx[r]);
            corr[r] = __expf(m[r] - mn);
            m[r] = mn;
        }
#pragma unroll
        for (int r = 0; r < 8; ++r) {
            const float p0 = __expf(s[0][r] - m[r]);
            const float p1 = __expf(s[1][r] - m[r]);
            psum[r] = p0 + p1;
            const int prow = (r + hi * 8) * 32;
            pw[prow + ln]      = (_Float16)p0;
            pw[prow + 16 + ln] = (_Float16)p1;
        }
#pragma unroll
        for (int off = 1; off < 16; off <<= 1)
#pragma unroll
            for (int r = 0; r < 8; ++r)
                psum[r] += __shfl_xor(psum[r], off, 32);
#pragma unroll
        for (int r = 0; r < 8; ++r) l[r] = l[r] * corr[r] + psum[r];

        // ---- rescale accumulator
#pragma unroll
        for (int vt = 0; vt < 8; ++vt)
#pragma unroll
            for (int r = 0; r < 8; ++r) o[vt][r] *= corr[r];

        // ---- P: C-layout -> A-layout via per-wave LDS bounce
        //      (DS ops are in-order within a wave; explicit split-counter wait)
        asm volatile("s_wait_dscnt 0" ::: "memory");
        v16h ap;
        {
            const _Float16* prow = pw + ln * 32;
            const int k0 = hi * 16;
            v8h p0 = *(const v8h*)(prow + k0);
            v8h p1 = *(const v8h*)(prow + k0 + 8);
#pragma unroll
            for (int e = 0; e < 8; ++e) { ap[e] = p0[e]; ap[8 + e] = p1[e]; }
        }

        // ---- O += P (16x32) x V (32x128): 8 WMMAs; V transposed => contiguous
#pragma unroll
        for (int vt = 0; vt < 8; ++vt) {
            const _Float16* __restrict__ vcol =
                Vb + (size_t)(vt * 16 + ln) * SS + kb;
            v16h bv = *(const v16h*)(vcol + hi * 16);
            o[vt] = __builtin_amdgcn_wmma_f32_16x16x32_f16(
                        false, ap, false, bv, (short)0, o[vt], false, false);
        }
    }

    // ---- epilogue: divide by l, write f32 output
    float rl[8];
#pragma unroll
    for (int r = 0; r < 8; ++r) rl[r] = 1.0f / l[r];
#pragma unroll
    for (int vt = 0; vt < 8; ++vt)
#pragma unroll
        for (int r = 0; r < 8; ++r) {
            const int row = qbase + r + hi * 8;
            out[(size_t)(b * SS + row) * DD + vt * 16 + ln] = o[vt][r] * rl[r];
        }
}

// ---------------------------------------------------------------------------
extern "C" void kernel_launch(void* const* d_in, const int* in_sizes, int n_in,
                              void* d_out, int out_size, void* d_ws, size_t ws_size,
                              hipStream_t stream) {
    (void)in_sizes; (void)n_in; (void)out_size; (void)ws_size;
    const float* X  = (const float*)d_in[0];
    const float* Wq = (const float*)d_in[1];
    const float* Wk = (const float*)d_in[2];
    const float* Wv = (const float*)d_in[3];
    float* out = (float*)d_out;

    const size_t qkv_elems = (size_t)BB * SS * DD;   // 2M f16 each
    _Float16* Qh = (_Float16*)d_ws;
    _Float16* Kh = Qh + qkv_elems;
    _Float16* Vt = Kh + qkv_elems;

    // 1024 row-blocks * 3 matrices = 3072 wave-tiles; 8 waves per block
    qkv_project_kernel<<<384, 256, 0, stream>>>(X, Wq, Wk, Wv, Qh, Kh, Vt);
    // 8 * 128 = 1024 q-tiles; 8 waves per block
    flash_attn_kernel<<<128, 256, 0, stream>>>(Qh, Kh, Vt, out);
}
